// MyLeNetMatStochBUNoceil_23940147708640
// MI455X (gfx1250) — matrix operand order repaired and checked by reading, not compiled, not run on hardware
//
#include <hip/hip_runtime.h>

typedef __attribute__((ext_vector_type(16))) __bf16 v16bf;
typedef __attribute__((ext_vector_type(8)))  __bf16 v8bf;
typedef __attribute__((ext_vector_type(8)))  float  v8f;

#define TM 64   // M tile (4 waves x 16 rows)

// gfx1250 async global->LDS staging via inline asm (bypasses builtin LangAS
// type-checking; ISA 15.18.3 op 98, VFLAT encoding, GV mode: saddr = off).
#if defined(__AMDGCN__)
#define ASYNC_B 1
#else
#define ASYNC_B 0
#endif

__device__ __forceinline__ void async_load_lds_16B(const void* gsrc, void* ldst) {
#if ASYNC_B
  unsigned ldsOff = (unsigned)(size_t)ldst;             // low 32b of generic = LDS byte addr
  unsigned long long ga = (unsigned long long)(size_t)gsrc;
  asm volatile("global_load_async_to_lds_b128 %0, %1, off"
               :: "v"(ldsOff), "v"(ga) : "memory");
#else
  *(v8bf*)ldst = *(const v8bf*)gsrc;
#endif
}

__device__ __forceinline__ void wait_asynccnt0() {
#if ASYNC_B
#if __has_builtin(__builtin_amdgcn_s_wait_asynccnt)
  __builtin_amdgcn_s_wait_asynccnt(0);
#else
  asm volatile("s_wait_asynccnt 0x0" ::: "memory");
#endif
#endif
}

// ---------------------------------------------------------------------------
// Pack fp32 NCHW input [64,3,32,32] -> bf16 channels-last [64,32,32,32] (pad 3->32)
// ---------------------------------------------------------------------------
__global__ void pack_input_kernel(const float* __restrict__ x, __bf16* __restrict__ act0) {
  int i = blockIdx.x * blockDim.x + threadIdx.x;   // pixel id over 64*32*32
  const int total = 64 * 32 * 32;
  if (i >= total) return;
  int b = i >> 10;          // /1024
  int rem = i & 1023;       // h*32+w
  __bf16* o = act0 + (size_t)i * 32;
  #pragma unroll
  for (int c = 0; c < 32; ++c) {
    float v = (c < 3) ? x[((size_t)b * 3 + c) * 1024 + rem] : 0.0f;
    o[c] = (__bf16)v;
  }
}

// ---------------------------------------------------------------------------
// Fused: 3x3 stride-1 pad-1 conv evaluated ONLY at stochastically gathered
// positions, + bias + ReLU, as implicit GEMM with bf16 WMMA.
//   D[Cpout, P] = W[Cpout, 9*Cpin] x patches[9*Cpin, P]
// Block = 128 threads (4 waves); tile = 64 M x TN N.
// All dims compile-time: Hin/OH are powers of two -> div/mod become shifts.
// ---------------------------------------------------------------------------
template<int Hin, int Cin, int Cpin, int Cout, int Cpout, int OH, int TN>
__global__ __launch_bounds__(128)
void sconv_wmma_kernel(const __bf16* __restrict__ actIn,
                       __bf16* __restrict__ actOut,
                       const float* __restrict__ w,
                       const float* __restrict__ bias,
                       const int* __restrict__ idx) {
  constexpr int NSUB = TN / 16;            // N subtiles per wave
  __shared__ __align__(16) __bf16 Ab[TM][32];   // weight tile  (M x K-chunk)
  __shared__ __align__(16) __bf16 Bb[TN][32];   // patch tile   (N x K-chunk)
  __shared__ int   sBase[TN];                   // input batch base offset
  __shared__ short sCy[TN], sCx[TN];            // gather center (afterconv grid)
  __shared__ int   sOut[TN];                    // output flat base offset

  const int tid    = threadIdx.x;
  const int lane   = tid & 31;
  const int wave   = tid >> 5;
  const int half   = lane >> 4;    // K-group selector for A / K-half for B
  const int l16    = lane & 15;    // M row (A) / N col (B,D)
  const int blockM = blockIdx.x * TM;
  const int blockN = blockIdx.y * TN;

  // Decode the TN gathered positions for this block (grid covers P exactly).
  for (int n = tid; n < TN; n += 128) {
    int p  = blockN + n;
    int b  = p / (OH * OH);                    // constant -> shift
    int r  = p % (OH * OH);
    int oy = r / OH, ox = r % OH;
    int fl = idx[p];                           // ih*Hin + iw (afterconv grid)
    sBase[n] = b * Hin * Hin * Cpin;
    sCy[n]   = (short)(fl / Hin);              // constant -> shift
    sCx[n]   = (short)(fl % Hin);
    sOut[n]  = ((b * OH + oy) * OH + ox) * Cpout;
  }
  __syncthreads();

  v8f acc[NSUB];
  #pragma unroll
  for (int j = 0; j < NSUB; ++j)
    #pragma unroll
    for (int e = 0; e < 8; ++e) acc[j][e] = 0.0f;

  constexpr int nChunk = Cpin / 32;
  #pragma unroll 1
  for (int tap = 0; tap < 9; ++tap) {
    const int dy = tap / 3 - 1, dx = tap % 3 - 1;
    #pragma unroll 1
    for (int ch = 0; ch < nChunk; ++ch) {
      const int c0 = ch * 32;

      // ---- stage B: TN positions x 32 channels, in 16B (8 x bf16) chunks ---
      #pragma unroll
      for (int q = tid; q < TN * 4; q += 128) {
        int n  = q >> 2;
        int k0 = (q & 3) << 3;                 // 8-element (16B) chunk
        int y = (int)sCy[n] + dy, x = (int)sCx[n] + dx;
        bool inb = ((unsigned)y < (unsigned)Hin) && ((unsigned)x < (unsigned)Hin);
        __bf16* dst = &Bb[n][k0];
        if (inb) {
          const __bf16* src = actIn + (size_t)sBase[n]
                              + (size_t)(y * Hin + x) * Cpin + c0 + k0;
          async_load_lds_16B(src, dst);        // global_load_async_to_lds_b128
        } else {
          v8bf z;
          #pragma unroll
          for (int e = 0; e < 8; ++e) z[e] = (__bf16)0.0f;
          *(v8bf*)dst = z;
        }
      }

      // ---- stage A: 64 rows x 32 k (fp32 -> bf16 on the fly; zero-padded) --
      {
        int mL = tid >> 1;
        int k0 = (tid & 1) << 4;
        int mG = blockM + mL;
        if (mG < Cout)   // prefetch next K-chunk of this weight row
          __builtin_prefetch(&w[((size_t)mG * Cin + (c0 + 32 + k0)) * 9 + tap], 0, 1);
        #pragma unroll
        for (int e = 0; e < 16; ++e) {
          int c = c0 + k0 + e;
          float v = (mG < Cout && c < Cin)
                        ? w[((size_t)mG * Cin + c) * 9 + tap] : 0.0f;
          Ab[mL][k0 + e] = (__bf16)v;
        }
      }
      wait_asynccnt0();                        // s_wait_asynccnt 0
      __syncthreads();

      // ---- A fragment (16x32): lane = M row, K = (e/8)*16 + half*8 + e%8 ---
      v16bf afrag;
      {
        const __bf16* ap = &Ab[(wave << 4) + l16][half << 3];
        #pragma unroll
        for (int e = 0; e < 8; ++e) { afrag[e] = ap[e]; afrag[e + 8] = ap[e + 16]; }
      }

      // ---- NSUB WMMAs: B fragment (32x16): lane%16 = N col, K = (e/8)*16 + half*8 + e%8 --
      #pragma unroll
      for (int j = 0; j < NSUB; ++j) {
        v16bf bfrag;
        const __bf16* bp = &Bb[(j << 4) + l16][half << 3];
        #pragma unroll
        for (int e = 0; e < 8; ++e) { bfrag[e] = bp[e]; bfrag[e + 8] = bp[e + 16]; }
        acc[j] = __builtin_amdgcn_wmma_f32_16x16x32_bf16(
            false, afrag, false, bfrag, (short)0, acc[j], false, false);
      }
      __syncthreads();
    }
  }

  // ---- epilogue: bias + ReLU, store bf16 channels-last at gathered position
  #pragma unroll
  for (int j = 0; j < NSUB; ++j) {
    int n = (j << 4) + l16;
    int obase = sOut[n];
    #pragma unroll
    for (int r = 0; r < 8; ++r) {
      int mL = (wave << 4) + r + (half << 3);   // D: M = r + 8*(lane/16)
      int mG = blockM + mL;
      if (mG < Cpout) {
        float bv = (mG < Cout) ? bias[mG] : 0.0f;
        float v  = acc[j][r] + bv;
        v = v > 0.0f ? v : 0.0f;
        actOut[(size_t)obase + mG] = (__bf16)v;
      }
    }
  }
}

// ---------------------------------------------------------------------------
// Final FC: out[64,10] = act4[64,4800](bf16) @ fc_w[10,4800]^T + fc_b
// ---------------------------------------------------------------------------
__global__ __launch_bounds__(256)
void fc_kernel(const __bf16* __restrict__ act, const float* __restrict__ w,
               const float* __restrict__ b, float* __restrict__ out) {
  __shared__ float red[256];
  const int bi = blockIdx.x;
  const __bf16* a = act + (size_t)bi * 4800;
  for (int j = 0; j < 10; ++j) {
    float s = 0.0f;
    for (int k = threadIdx.x; k < 4800; k += 256)
      s += (float)a[k] * w[j * 4800 + k];
    red[threadIdx.x] = s;
    __syncthreads();
    for (int off = 128; off > 0; off >>= 1) {
      if (threadIdx.x < off) red[threadIdx.x] += red[threadIdx.x + off];
      __syncthreads();
    }
    if (threadIdx.x == 0) out[bi * 10 + j] = red[0] + b[j];
    __syncthreads();
  }
}

// ---------------------------------------------------------------------------
extern "C" void kernel_launch(void* const* d_in, const int* in_sizes, int n_in,
                              void* d_out, int out_size, void* d_ws, size_t ws_size,
                              hipStream_t stream) {
  // Input order (setup_inputs dict order):
  // 0:x 1:w1 2:b1 3:index1 4:w2 5:b2 6:index2 7:w3 8:b3 9:index3
  // 10:w4 11:b4 12:index4 13:fc_w 14:fc_b
  const float* x   = (const float*)d_in[0];
  const float* w1  = (const float*)d_in[1];
  const float* b1  = (const float*)d_in[2];
  const int*   i1  = (const int*)d_in[3];
  const float* w2  = (const float*)d_in[4];
  const float* b2  = (const float*)d_in[5];
  const int*   i2  = (const int*)d_in[6];
  const float* w3  = (const float*)d_in[7];
  const float* b3  = (const float*)d_in[8];
  const int*   i3  = (const int*)d_in[9];
  const float* w4  = (const float*)d_in[10];
  const float* b4  = (const float*)d_in[11];
  const int*   i4  = (const int*)d_in[12];
  const float* fcw = (const float*)d_in[13];
  const float* fcb = (const float*)d_in[14];
  float* out = (float*)d_out;

  // Workspace: bf16 channels-last activations (channels padded to mult. of 32)
  __bf16* act0 = (__bf16*)d_ws;                       // [64,32,32,  32]  4.0 MB
  __bf16* act1 = act0 + (size_t)64 * 1024 * 32;       // [64,16,16, 608] 19.9 MB
  __bf16* act2 = act1 + (size_t)64 * 256  * 608;      // [64, 8, 8,1216] 10.0 MB
  __bf16* act3 = act2 + (size_t)64 * 64   * 1216;     // [64, 4, 4,2400]  4.9 MB
  __bf16* act4 = act3 + (size_t)64 * 16   * 2400;     // [64, 1, 1,4800]  0.6 MB

  pack_input_kernel<<<(64 * 1024 + 255) / 256, 256, 0, stream>>>(x, act0);

  // L1: Hin=32 Cin=3    Cpin=32   Cout=600  Cpout=608  OH=16  P=16384, TN=128
  sconv_wmma_kernel<32, 3, 32, 600, 608, 16, 128>
      <<<dim3(10, 128), 128, 0, stream>>>(act0, act1, w1, b1, i1);
  // L2: Hin=16 Cin=600  Cpin=608  Cout=1200 Cpout=1216 OH=8   P=4096,  TN=128
  sconv_wmma_kernel<16, 600, 608, 1200, 1216, 8, 128>
      <<<dim3(19, 32), 128, 0, stream>>>(act1, act2, w2, b2, i2);
  // L3: Hin=8  Cin=1200 Cpin=1216 Cout=2400 Cpout=2400 OH=4   P=1024,  TN=128
  sconv_wmma_kernel<8, 1200, 1216, 2400, 2400, 4, 128>
      <<<dim3(38, 8), 128, 0, stream>>>(act2, act3, w3, b3, i3);
  // L4: Hin=4  Cin=2400 Cpin=2400 Cout=4800 Cpout=4800 OH=1   P=64,    TN=64
  sconv_wmma_kernel<4, 2400, 2400, 4800, 4800, 1, 64>
      <<<dim3(75, 1), 128, 0, stream>>>(act3, act4, w4, b4, i4);

  fc_kernel<<<64, 256, 0, stream>>>(act4, fcw, fcb, out);
  (void)in_sizes; (void)n_in; (void)out_size; (void)ws_size;
}
